// GPT_78735340470777
// MI455X (gfx1250) — matrix operand order repaired and checked by reading, not compiled, not run on hardware
//
#include <hip/hip_runtime.h>
#include <hip/hip_bf16.h>
#include <math.h>
#include <stdint.h>

#define D_  1024
#define H_  8
#define HD_ 128
#define L_  4
#define V_  50257
#define T_  2048

typedef __bf16 bf16;
typedef __attribute__((ext_vector_type(16))) __bf16 v16bf;
typedef __attribute__((ext_vector_type(8)))  __bf16 v8bf;
typedef __attribute__((ext_vector_type(8)))  float  v8f;
typedef int v4i __attribute__((vector_size(16)));
typedef __attribute__((address_space(3))) v4i* lds_v4i_ptr;

static __device__ __forceinline__ v16bf cat8(v8bf lo, v8bf hi) {
  return __builtin_shufflevector(lo, hi, 0,1,2,3,4,5,6,7,8,9,10,11,12,13,14,15);
}

// ---------------------------------------------------------------------------
// CDNA5 async global->LDS path (ASYNCcnt-tracked DMA, bypasses VGPRs).
// Probe results: builtin signature is (v4i* generic, v4i addrspace(3)*, imm, imm).
// Guarded: falls back to synchronous vector copies if the builtin is absent.
// ---------------------------------------------------------------------------
#if defined(__has_builtin)
#if __has_builtin(__builtin_amdgcn_global_load_async_to_lds_b128)
#define HAVE_ASYNC 1
#endif
#endif
#ifndef HAVE_ASYNC
#define HAVE_ASYNC 0
#endif

#define TO_GLB(p) ((v4i*)(uintptr_t)(p))
#define TO_LDS(p) ((lds_v4i_ptr)((__attribute__((address_space(3))) void*)(void*)(p)))

template<int N>
static __device__ __forceinline__ void async_wait() {
#if HAVE_ASYNC
#if __has_builtin(__builtin_amdgcn_s_wait_asynccnt)
  __builtin_amdgcn_s_wait_asynccnt((unsigned short)N);
#else
  asm volatile("s_wait_asynccnt %0" :: "i"(N) : "memory");
#endif
#endif
}

static __device__ __forceinline__ void async_cp32(const bf16* g, bf16* l) {
#if HAVE_ASYNC
  __builtin_amdgcn_global_load_async_to_lds_b128(TO_GLB(g),     TO_LDS(l),     0, 0);
  __builtin_amdgcn_global_load_async_to_lds_b128(TO_GLB(g + 8), TO_LDS(l + 8), 0, 0);
#else
  *(v16bf*)l = *(const v16bf*)g;
#endif
}

// ---------------------------------------------------------------------------
// GEMM (B fp32, converted inline): C[M,N] (+)= scale * A_bf16[M,K] x B[N,K]^T
// EPI: 0 = store fp32, 1 = add fp32 (residual), 2 = relu^2 -> bf16 store
// 128x128 tile, BK=32, 8 waves (4M x 2N), double-buffered; A tile via async DMA.
// ---------------------------------------------------------------------------
template<int EPI>
__global__ __launch_bounds__(256)
void gemm_bf16_kernel(const bf16* __restrict__ A, const float* __restrict__ B,
                      float* __restrict__ Cf, bf16* __restrict__ Cb,
                      int M, int N, int K,
                      const float* __restrict__ scale_ptr, int scale_idx)
{
  __shared__ alignas(64) bf16 sA[2][128][32];
  __shared__ alignas(64) bf16 sB[2][128][32];
  const int tid  = threadIdx.x;
  const int wave = tid >> 5, lane = tid & 31;
  const int hl   = lane >> 4, ln = lane & 15;
  const int mw   = wave & 3,  nw = wave >> 2;
  const int m0   = blockIdx.y * 128, n0 = blockIdx.x * 128;
  const int lr   = tid >> 1, lh = tid & 1;
  (void)M;

  v8f acc[2][4] = {};
  float br[16];

  auto issueA = [&](int buf, int kt) {
    async_cp32(A + (size_t)(m0 + lr) * K + kt + lh * 16, &sA[buf][lr][lh * 16]);
  };
  auto loadB = [&](int kt) {
    const int n = n0 + lr;
    if (n < N) {
      const float* bp = B + (size_t)n * K + kt + lh * 16;
      #pragma unroll
      for (int j = 0; j < 16; ++j) br[j] = bp[j];
    } else {
      #pragma unroll
      for (int j = 0; j < 16; ++j) br[j] = 0.0f;
    }
  };

  issueA(0, 0);
  loadB(0);
  int cur = 0;
  for (int kt = 0; kt < K; kt += 32) {
    {
      v16bf bb;
      #pragma unroll
      for (int j = 0; j < 16; ++j) bb[j] = (bf16)br[j];
      *(v16bf*)&sB[cur][lr][lh * 16] = bb;
    }
    const bool more = (kt + 32) < K;
    if (more) { issueA(cur ^ 1, kt + 32); loadB(kt + 32); }
    if (more) async_wait<2>(); else async_wait<0>();
    __syncthreads();

    v16bf a[2];
    #pragma unroll
    for (int mf = 0; mf < 2; ++mf) {
      const int row = mw * 32 + mf * 16 + ln;
      const int kc  = hl * 8;
      a[mf] = cat8(*(const v8bf*)&sA[cur][row][kc], *(const v8bf*)&sA[cur][row][kc + 16]);
    }
    #pragma unroll
    for (int nf = 0; nf < 4; ++nf) {
      const int brow = nw * 64 + nf * 16 + ln;
      v16bf b = cat8(*(const v8bf*)&sB[cur][brow][hl * 8], *(const v8bf*)&sB[cur][brow][hl * 8 + 16]);
      #pragma unroll
      for (int mf = 0; mf < 2; ++mf)
        acc[mf][nf] = __builtin_amdgcn_wmma_f32_16x16x32_bf16(
            false, a[mf], false, b, (short)0, acc[mf][nf], false, false);
    }
    __syncthreads();
    cur ^= 1;
  }

  const float s = scale_ptr ? scale_ptr[scale_idx] : 1.0f;
  #pragma unroll
  for (int mf = 0; mf < 2; ++mf)
    #pragma unroll
    for (int nf = 0; nf < 4; ++nf)
      #pragma unroll
      for (int r = 0; r < 8; ++r) {
        const int gm = m0 + mw * 32 + mf * 16 + r + 8 * hl;
        const int gn = n0 + nw * 64 + nf * 16 + ln;
        if (gn < N) {
          const float val = acc[mf][nf][r] * s;
          const size_t o = (size_t)gm * N + gn;
          if (EPI == 0)      Cf[o] = val;
          else if (EPI == 1) Cf[o] += val;
          else { const float rr = val > 0.f ? val : 0.f; Cb[o] = (bf16)(rr * rr); }
        }
      }
}

// ---------------------------------------------------------------------------
// All-bf16 GEMM (lm_head): both tiles via async DMA, double-buffered.
// B must be addressable for ceil(N/128)*128 rows (padded buffer).
// ---------------------------------------------------------------------------
__global__ __launch_bounds__(256)
void gemm_bb_kernel(const bf16* __restrict__ A, const bf16* __restrict__ B,
                    float* __restrict__ Cf, int N, int K)
{
  __shared__ alignas(64) bf16 sA[2][128][32];
  __shared__ alignas(64) bf16 sB[2][128][32];
  const int tid  = threadIdx.x;
  const int wave = tid >> 5, lane = tid & 31;
  const int hl   = lane >> 4, ln = lane & 15;
  const int mw   = wave & 3,  nw = wave >> 2;
  const int m0   = blockIdx.y * 128, n0 = blockIdx.x * 128;
  const int lr   = tid >> 1, lh = tid & 1;

  v8f acc[2][4] = {};

  auto issue = [&](int buf, int kt) {
    async_cp32(A + (size_t)(m0 + lr) * K + kt + lh * 16, &sA[buf][lr][lh * 16]);
    async_cp32(B + (size_t)(n0 + lr) * K + kt + lh * 16, &sB[buf][lr][lh * 16]);
  };

  issue(0, 0);
  int cur = 0;
  for (int kt = 0; kt < K; kt += 32) {
    const bool more = (kt + 32) < K;
    if (more) issue(cur ^ 1, kt + 32);
    if (more) async_wait<4>(); else async_wait<0>();
    __syncthreads();

    v16bf a[2];
    #pragma unroll
    for (int mf = 0; mf < 2; ++mf) {
      const int row = mw * 32 + mf * 16 + ln;
      const int kc  = hl * 8;
      a[mf] = cat8(*(const v8bf*)&sA[cur][row][kc], *(const v8bf*)&sA[cur][row][kc + 16]);
    }
    #pragma unroll
    for (int nf = 0; nf < 4; ++nf) {
      const int brow = nw * 64 + nf * 16 + ln;
      v16bf b = cat8(*(const v8bf*)&sB[cur][brow][hl * 8], *(const v8bf*)&sB[cur][brow][hl * 8 + 16]);
      #pragma unroll
      for (int mf = 0; mf < 2; ++mf)
        acc[mf][nf] = __builtin_amdgcn_wmma_f32_16x16x32_bf16(
            false, a[mf], false, b, (short)0, acc[mf][nf], false, false);
    }
    __syncthreads();
    cur ^= 1;
  }

  #pragma unroll
  for (int mf = 0; mf < 2; ++mf)
    #pragma unroll
    for (int nf = 0; nf < 4; ++nf)
      #pragma unroll
      for (int r = 0; r < 8; ++r) {
        const int gm = m0 + mw * 32 + mf * 16 + r + 8 * hl;
        const int gn = n0 + nw * 64 + nf * 16 + ln;
        if (gn < N) Cf[(size_t)gm * N + gn] = acc[mf][nf][r];
      }
}

// ---------------------------------------------------------------------------
// Flash attention: one (64-row q block, head) per 128-thread block (4 waves).
// Q/K tiles via async DMA; S = Q K^T (16 WMMA), online softmax, O += P V.
// ---------------------------------------------------------------------------
__global__ __launch_bounds__(128)
void flash_kernel(const bf16* __restrict__ q, const bf16* __restrict__ k,
                  const bf16* __restrict__ v, float* __restrict__ o)
{
  const int qbk = blockIdx.x, h = blockIdx.y;
  const int tid = threadIdx.x, wave = tid >> 5, lane = tid & 31;
  const int hl = lane >> 4, ln = lane & 15;

  __shared__ alignas(64) bf16 sQ[64][128];
  __shared__ alignas(64) bf16 sK[64][128];
  __shared__ alignas(64) bf16 sV[128][64];     // transposed: [hd][s]
  __shared__ alignas(64) bf16 sP[4][16][64];   // per-wave P tile

  const bf16* qh = q + ((size_t)h * T_ + (size_t)qbk * 64) * HD_;
  for (int i = tid; i < 64 * 8; i += 128) {
    const int row = i >> 3, c = (i & 7) << 4;
    async_cp32(qh + (size_t)row * HD_ + c, &sQ[row][c]);
  }
  async_wait<0>();
  __syncthreads();

  v16bf aq[4];
  #pragma unroll
  for (int ks = 0; ks < 4; ++ks) {
    const int kc = ks * 32 + hl * 8;
    aq[ks] = cat8(*(const v8bf*)&sQ[wave * 16 + ln][kc],
                  *(const v8bf*)&sQ[wave * 16 + ln][kc + 16]);
  }

  float m_i[8], l_i[8];
  v8f oacc[8] = {};
  #pragma unroll
  for (int r = 0; r < 8; ++r) { m_i[r] = -1e30f; l_i[r] = 0.f; }

  for (int kbk = 0; kbk <= qbk; ++kbk) {
    __syncthreads();
    const bf16* kh = k + ((size_t)h * T_ + (size_t)kbk * 64) * HD_;
    for (int i = tid; i < 64 * 8; i += 128) {
      const int row = i >> 3, c = (i & 7) << 4;
      async_cp32(kh + (size_t)row * HD_ + c, &sK[row][c]);
    }
    const bf16* vh = v + ((size_t)h * T_ + (size_t)kbk * 64) * HD_;
    for (int i = tid; i < 64 * 128; i += 128) {
      const int srow = i >> 7, hd = i & 127;
      sV[hd][srow] = vh[(size_t)srow * HD_ + hd];
    }
    async_wait<0>();
    __syncthreads();

    // --- S = Q K^T ---
    v8f sc[4] = {};
    #pragma unroll
    for (int nf = 0; nf < 4; ++nf) {
      const int krow = nf * 16 + ln;
      #pragma unroll
      for (int ks = 0; ks < 4; ++ks) {
        v16bf b = cat8(*(const v8bf*)&sK[krow][ks * 32 + hl * 8], *(const v8bf*)&sK[krow][ks * 32 + hl * 8 + 16]);
        sc[nf] = __builtin_amdgcn_wmma_f32_16x16x32_bf16(
            false, aq[ks], false, b, (short)0, sc[nf], false, false);
      }
    }

    // --- online softmax (rows live in VGPR index r + half-wave) ---
    #pragma unroll
    for (int r = 0; r < 8; ++r) {
      const int trow = qbk * 64 + wave * 16 + r + 8 * hl;
      float rmax = -1e30f;
      #pragma unroll
      for (int nf = 0; nf < 4; ++nf) {
        const int scol = kbk * 64 + nf * 16 + ln;
        float val = sc[nf][r] * 0.1f;            // ATTN_SCALE
        if (scol > trow) val = -1e30f;           // causal mask
        sc[nf][r] = val;
        rmax = fmaxf(rmax, val);
      }
      #pragma unroll
      for (int off = 1; off < 16; off <<= 1)
        rmax = fmaxf(rmax, __shfl_xor(rmax, off, 32));
      const float mnew = fmaxf(m_i[r], rmax);
      const float corr = expf(m_i[r] - mnew);
      float rsum = 0.f;
      #pragma unroll
      for (int nf = 0; nf < 4; ++nf) {
        const float pv = expf(sc[nf][r] - mnew);
        sc[nf][r] = pv;
        rsum += pv;
      }
      #pragma unroll
      for (int off = 1; off < 16; off <<= 1) rsum += __shfl_xor(rsum, off, 32);
      l_i[r] = l_i[r] * corr + rsum;
      m_i[r] = mnew;
      #pragma unroll
      for (int nf = 0; nf < 8; ++nf) oacc[nf][r] *= corr;
    }

    // --- P through LDS to re-layout C -> A ---
    #pragma unroll
    for (int nf = 0; nf < 4; ++nf)
      #pragma unroll
      for (int r = 0; r < 8; ++r)
        sP[wave][r + 8 * hl][nf * 16 + ln] = (bf16)sc[nf][r];
    __syncthreads();

    v16bf ap[2];
    #pragma unroll
    for (int ks = 0; ks < 2; ++ks) {
      const int kc = ks * 32 + hl * 8;
      ap[ks] = cat8(*(const v8bf*)&sP[wave][ln][kc],
                    *(const v8bf*)&sP[wave][ln][kc + 16]);
    }
    #pragma unroll
    for (int nf = 0; nf < 8; ++nf) {
      const int vrow = nf * 16 + ln;
      #pragma unroll
      for (int ks = 0; ks < 2; ++ks) {
        v16bf b = cat8(*(const v8bf*)&sV[vrow][ks * 32 + hl * 8], *(const v8bf*)&sV[vrow][ks * 32 + hl * 8 + 16]);
        oacc[nf] = __builtin_amdgcn_wmma_f32_16x16x32_bf16(
            false, ap[ks], false, b, (short)0, oacc[nf], false, false);
      }
    }
  }

  #pragma unroll
  for (int r = 0; r < 8; ++r) {
    const float inv = 1.0f / l_i[r];
    const int trow = qbk * 64 + wave * 16 + r + 8 * hl;
    #pragma unroll
    for (int nf = 0; nf < 8; ++nf)
      o[(size_t)trow * D_ + h * HD_ + nf * 16 + ln] = oacc[nf][r] * inv;
  }
}

// ---------------------------------------------------------------------------
// Elementwise / reduction kernels
// ---------------------------------------------------------------------------
__global__ __launch_bounds__(256)
void f32_to_bf16_kernel(const float* __restrict__ in, bf16* __restrict__ out, long n)
{
  const long base = (long)blockIdx.x * 2048 + threadIdx.x;
  #pragma unroll
  for (int j = 0; j < 8; ++j) {
    const long p = base + (long)j * 256;
    if (p < n) out[p] = (bf16)in[p];
  }
}

__global__ __launch_bounds__(256)
void embed_rms_kernel(const int* __restrict__ idx, const float* __restrict__ embed,
                      float* __restrict__ x)
{
  const int t = blockIdx.x, tid = threadIdx.x;
  const int tok = idx[t];
  const float* er = embed + (size_t)tok * D_;
  float v[4]; float ss = 0.f;
  #pragma unroll
  for (int i = 0; i < 4; ++i) { v[i] = er[tid + i * 256]; ss += v[i] * v[i]; }
  #pragma unroll
  for (int off = 1; off < 32; off <<= 1) ss += __shfl_xor(ss, off, 32);
  __shared__ float red[8];
  if ((tid & 31) == 0) red[tid >> 5] = ss;
  __syncthreads();
  float tot = 0.f;
  #pragma unroll
  for (int w = 0; w < 8; ++w) tot += red[w];
  const float sc = rsqrtf(tot / (float)D_ + 1e-6f);
  float* xr = x + (size_t)t * D_;
  #pragma unroll
  for (int i = 0; i < 4; ++i) xr[tid + i * 256] = v[i] * sc;
}

__global__ __launch_bounds__(256)
void rmsnorm_kernel(const float* __restrict__ x, bf16* __restrict__ out)
{
  const int t = blockIdx.x, tid = threadIdx.x;
  const float* xr = x + (size_t)t * D_;
  float v[4]; float ss = 0.f;
  #pragma unroll
  for (int i = 0; i < 4; ++i) { v[i] = xr[tid + i * 256]; ss += v[i] * v[i]; }
  #pragma unroll
  for (int off = 1; off < 32; off <<= 1) ss += __shfl_xor(ss, off, 32);
  __shared__ float red[8];
  if ((tid & 31) == 0) red[tid >> 5] = ss;
  __syncthreads();
  float tot = 0.f;
  #pragma unroll
  for (int w = 0; w < 8; ++w) tot += red[w];
  const float sc = rsqrtf(tot / (float)D_ + 1e-6f);
  bf16* orow = out + (size_t)t * D_;
  #pragma unroll
  for (int i = 0; i < 4; ++i) orow[tid + i * 256] = (bf16)(v[i] * sc);
}

// Per-(token, head-slot): head RMSNorm + YaRN rotary for q/k, copy for v; -> bf16 [H][T][HD]
__global__ __launch_bounds__(128)
void rope_kernel(const float* __restrict__ qkv,
                 bf16* __restrict__ q, bf16* __restrict__ k, bf16* __restrict__ v)
{
  const int t = blockIdx.x, slot = blockIdx.y, d = threadIdx.x; // slot: 0..23
  float val = qkv[(size_t)t * (3 * D_) + (size_t)slot * HD_ + d];
  const int kind = slot >> 3, h = slot & 7;
  if (kind < 2) {
    float ss = val * val;
    #pragma unroll
    for (int off = 1; off < 32; off <<= 1) ss += __shfl_xor(ss, off, 32);
    __shared__ float red[4];
    if ((d & 31) == 0) red[d >> 5] = ss;
    __syncthreads();
    const float tot = red[0] + red[1] + red[2] + red[3];
    val *= rsqrtf(tot / (float)HD_ + 1e-6f);
    const float partner = __shfl_xor(val, 1, 32);  // pairwise flip
    const float af = (d < 64) ? powf(1.0f / 1024.0f, (float)(d >> 1) * (1.0f / 31.0f)) : 0.0f;
    const float th = (float)t * af;
    const float c = cosf(th), sn = sinf(th);
    const float f2 = (d & 1) ? -sn : sn;
    val = c * val + f2 * partner;
  }
  const size_t o = ((size_t)h * T_ + t) * HD_ + d;
  const bf16 bv = (bf16)val;
  if (kind == 0) q[o] = bv; else if (kind == 1) k[o] = bv; else v[o] = bv;
}

// gate = 2*sigmoid(xn . gw[h]); og = (o * gate) as bf16, one wave per head
__global__ __launch_bounds__(256)
void gate_kernel(const bf16* __restrict__ xn, const float* __restrict__ gw,
                 const float* __restrict__ ob, bf16* __restrict__ og)
{
  const int t = blockIdx.x, tid = threadIdx.x;
  const int h = tid >> 5, lane = tid & 31;
  const bf16* xr = xn + (size_t)t * D_;
  const float* gr = gw + (size_t)h * D_;
  float dot = 0.f;
  #pragma unroll
  for (int i = 0; i < 32; ++i) {
    const int d = lane + i * 32;
    dot += (float)xr[d] * gr[d];
  }
  #pragma unroll
  for (int off = 1; off < 32; off <<= 1) dot += __shfl_xor(dot, off, 32);
  const float g = 2.0f / (1.0f + expf(-dot));
  #pragma unroll
  for (int e = 0; e < 4; ++e) {
    const int hd = lane + e * 32;
    const size_t o = (size_t)t * D_ + h * HD_ + hd;
    og[o] = (bf16)(ob[o] * g);
  }
}

// ---------------------------------------------------------------------------
// Host side
// ---------------------------------------------------------------------------
static void gemm_launch(int epi, const bf16* A, const float* B, float* Cf, bf16* Cb,
                        int M, int N, int K, const float* sp, int si, hipStream_t st)
{
  dim3 g((N + 127) / 128, M / 128);
  switch (epi) {
    case 0:  gemm_bf16_kernel<0><<<g, 256, 0, st>>>(A, B, Cf, Cb, M, N, K, sp, si); break;
    case 1:  gemm_bf16_kernel<1><<<g, 256, 0, st>>>(A, B, Cf, Cb, M, N, K, sp, si); break;
    default: gemm_bf16_kernel<2><<<g, 256, 0, st>>>(A, B, Cf, Cb, M, N, K, sp, si); break;
  }
}

extern "C" void kernel_launch(void* const* d_in, const int* in_sizes, int n_in,
                              void* d_out, int out_size, void* d_ws, size_t ws_size,
                              hipStream_t stream)
{
  (void)in_sizes; (void)n_in; (void)out_size; (void)ws_size;
  const int*   idx     = (const int*)  d_in[0];
  const float* embed   = (const float*)d_in[1];
  const float* qkvo_w  = (const float*)d_in[2];
  const float* lambdas = (const float*)d_in[3];
  const float* gate_w  = (const float*)d_in[4];
  const float* fc_w    = (const float*)d_in[5];
  const float* proj_w  = (const float*)d_in[6];
  const float* lm_head = (const float*)d_in[7];
  float* out = (float*)d_out;

  // Workspace carve-out (~180 MB total)
  char* p = (char*)d_ws;
  auto take = [&](size_t bytes) -> char* {
    char* r = p; p += (bytes + 255) & ~(size_t)255; return r;
  };
  const size_t NPAD = (size_t)((V_ + 127) / 128) * 128;  // padded lm_head rows
  float* x   = (float*)take((size_t)T_ * D_ * 4);        // residual stream fp32
  bf16*  xn  = (bf16*) take((size_t)T_ * D_ * 2);        // rmsnorm(x) bf16
  float* qkv = (float*)take((size_t)T_ * 3 * D_ * 4);    // qkv projections fp32
  bf16*  qb  = (bf16*) take((size_t)H_ * T_ * HD_ * 2);  // rotated q bf16
  bf16*  kb  = (bf16*) take((size_t)H_ * T_ * HD_ * 2);  // rotated k bf16
  bf16*  vb  = (bf16*) take((size_t)H_ * T_ * HD_ * 2);  // v bf16
  float* ob  = (float*)take((size_t)T_ * D_ * 4);        // attention out fp32
  bf16*  og  = (bf16*) take((size_t)T_ * D_ * 2);        // gated attention bf16
  bf16*  hb  = (bf16*) take((size_t)T_ * 4 * D_ * 2);    // relu^2 MLP hidden bf16
  bf16*  lmb = (bf16*) take(NPAD * (size_t)D_ * 2);      // lm_head bf16 (L2-resident)

  // One-time (per call) lm_head fp32 -> bf16: 206 MB (> L2) becomes 103 MB (< 192 MB L2)
  {
    const long n = (long)V_ * D_;
    const int grid = (int)((n + 2047) / 2048);
    f32_to_bf16_kernel<<<grid, 256, 0, stream>>>(lm_head, lmb, n);
  }

  embed_rms_kernel<<<T_, 256, 0, stream>>>(idx, embed, x);

  for (int l = 0; l < L_; ++l) {
    const float* wqkvo = qkvo_w + (size_t)l * 4 * D_ * D_;
    rmsnorm_kernel<<<T_, 256, 0, stream>>>(x, xn);
    // qkv = lambda0 * xn @ Wqkv^T
    gemm_launch(0, xn, wqkvo, qkv, nullptr, T_, 3 * D_, D_, lambdas, 2 * l + 0, stream);
    rope_kernel<<<dim3(T_, 3 * H_), 128, 0, stream>>>(qkv, qb, kb, vb);
    flash_kernel<<<dim3(T_ / 64, H_), 128, 0, stream>>>(qb, kb, vb, ob);
    gate_kernel<<<T_, 256, 0, stream>>>(xn, gate_w + (size_t)l * H_ * D_, ob, og);
    // x += lambda1 * og @ Wo^T
    gemm_launch(1, og, wqkvo + (size_t)3 * D_ * D_, x, nullptr, T_, D_, D_,
                lambdas, 2 * l + 1, stream);
    rmsnorm_kernel<<<T_, 256, 0, stream>>>(x, xn);
    // h = relu(xn @ Wfc^T)^2
    gemm_launch(2, xn, fc_w + (size_t)l * 4 * D_ * D_, nullptr, hb, T_, 4 * D_, D_,
                nullptr, 0, stream);
    // x += h @ Wproj^T
    gemm_launch(1, hb, proj_w + (size_t)l * D_ * 4 * D_, x, nullptr, T_, D_, 4 * D_,
                nullptr, 0, stream);
  }

  rmsnorm_kernel<<<T_, 256, 0, stream>>>(x, xn);
  // logits = xn @ lm_head^T  (all-bf16, async double-buffered)
  gemm_bb_kernel<<<dim3((V_ + 127) / 128, T_ / 128), 256, 0, stream>>>(
      xn, lmb, out, V_, D_);
}
